// UNetV2FPNMulti_81982335746519
// MI455X (gfx1250) — matrix operand order repaired and checked by reading, not compiled, not run on hardware
//
#include <hip/hip_runtime.h>
#include <hip/hip_bf16.h>

typedef __attribute__((ext_vector_type(16))) _Float16 v16h;
typedef __attribute__((ext_vector_type(8)))  float    v8f;

#define BN_EPS 1e-3f

__host__ __device__ constexpr int ctz_c(int v) {
  int c = 0;
  while (!(v & 1)) { v >>= 1; ++c; }
  return c;
}

// ---------------------------------------------------------------------------
// Implicit-GEMM conv (forward strided OR transposed/lhs-dilated) on WMMA
// f32_16x16x32_f16. One wave computes an (MV*16 voxel) x (NCO*16 cout) tile:
// MV*NCO WMMAs per K-chunk, so the expensive A-gather is amortized over NCO
// and the weight load over MV. Within a lane's staged A elements, channels are
// memory-contiguous in groups of G=min(CIN,16) sharing one kernel tap, so tap
// decode + bounds check happen once per group and data moves as float4 (b128).
// Weights (DHWIO) flatten so contraction index kk addresses w at kk*cout+co.
// ---------------------------------------------------------------------------
template <int CIN, int KD, int KH, int KW, int SD, int SH, int SW, int TR,
          int MV, int NCO>
__global__ __launch_bounds__(32)
void conv_wmma_kernel(const float* __restrict__ x, const float* __restrict__ w,
                      const float* __restrict__ mask, float* __restrict__ y,
                      int Di, int Hi, int Wi, int Do_, int Ho, int Wo,
                      int cout, int pd, int ph, int pw) {
  constexpr int KTOT = KD * KH * KW * CIN;
  constexpr int CSH = ctz_c(CIN);
  constexpr int G  = (CIN < 16) ? CIN : 16;  // contiguous channels per tap
  constexpr int NG = 16 / G;                 // groups per lane per chunk

  const int lane  = threadIdx.x;
  const int nvox  = Do_ * Ho * Wo;
  const int vbase = blockIdx.x * (MV * 16);
  const int co0   = blockIdx.y * (NCO * 16);

  __shared__ _Float16 lA[MV * 16][32];
  __shared__ _Float16 lB[32][NCO * 16];

  // A staging: lane covers row mA of each voxel tile, K-half (lane&1)*16..+15
  const int mA     = lane >> 1;
  const int kAbase = (lane & 1) * 16;

  int od[MV], oh[MV], ow[MV];
  bool vvalid[MV];
#pragma unroll
  for (int mv = 0; mv < MV; ++mv) {
    const int vo = vbase + mv * 16 + mA;
    vvalid[mv] = vo < nvox;
    if (vvalid[mv]) {
      ow[mv] = vo % Wo;
      oh[mv] = (vo / Wo) % Ho;
      od[mv] = vo / (Wo * Ho);
    } else {
      ow[mv] = oh[mv] = od[mv] = 0;
    }
  }

  v8f acc[MV][NCO];
#pragma unroll
  for (int mv = 0; mv < MV; ++mv)
#pragma unroll
    for (int t = 0; t < NCO; ++t) {
      v8f z = {};
      acc[mv][t] = z;
    }

  for (int k0 = 0; k0 < KTOT; k0 += 32) {
    // keep the next weight K-chunk warm (global_prefetch_b8)
    if (k0 + 32 < KTOT)
      __builtin_prefetch(&w[(size_t)(k0 + 32 + lane) * cout + co0], 0, 1);

    // ---- stage A tiles: (MV*16 voxels) x 32 contraction elems ----
#pragma unroll
    for (int mv = 0; mv < MV; ++mv) {
#pragma unroll
      for (int g = 0; g < NG; ++g) {
        const int kk0 = k0 + kAbase + g * G;  // group start, multiple of G
        bool ok = vvalid[mv] && ((KTOT % 32 == 0) || kk0 < KTOT);
        size_t base = 0;
        if (ok) {
          const int tap = kk0 >> CSH;
          const int ci0 = kk0 & (CIN - 1);
          const int tw  = tap % KW;
          const int th  = (tap / KW) % KH;
          const int td  = tap / (KW * KH);
          int id, ih, iw;
          if (!TR) {
            id = od[mv] * SD + td - pd;
            ih = oh[mv] * SH + th - ph;
            iw = ow[mv] * SW + tw - pw;
          } else {
            // transposed conv: i = (o + p - t) / s when divisible
            const int nd = od[mv] + pd - td;
            const int nh = oh[mv] + ph - th;
            const int nw = ow[mv] + pw - tw;
            ok = ok && (nd >= 0) && (nh >= 0) && (nw >= 0) &&
                 (nd % SD == 0) && (nh % SH == 0) && (nw % SW == 0);
            id = nd / SD; ih = nh / SH; iw = nw / SW;
          }
          ok = ok && ((unsigned)id < (unsigned)Di) &&
                     ((unsigned)ih < (unsigned)Hi) &&
                     ((unsigned)iw < (unsigned)Wi);
          base = ((((size_t)id * Hi + ih) * Wi + iw) << CSH) + ci0;
        }
#pragma unroll
        for (int j = 0; j < G; j += 4) {
          const float4 v4 =
              ok ? *(const float4*)&x[base + j] : make_float4(0, 0, 0, 0);
          lA[mv * 16 + mA][kAbase + g * G + j + 0] = (_Float16)v4.x;
          lA[mv * 16 + mA][kAbase + g * G + j + 1] = (_Float16)v4.y;
          lA[mv * 16 + mA][kAbase + g * G + j + 2] = (_Float16)v4.z;
          lA[mv * 16 + mA][kAbase + g * G + j + 3] = (_Float16)v4.w;
        }
      }
    }
    // ---- stage B tile: 32 contraction elems x NCO*16 couts (contiguous) ----
    {
      const int kk = k0 + lane;
      const bool kv = (KTOT % 32 == 0) || kk < KTOT;
      const float* wrow = &w[(size_t)kk * cout + co0];
#pragma unroll
      for (int j = 0; j < NCO * 16; j += 4) {
        const float4 v4 =
            kv ? *(const float4*)(wrow + j) : make_float4(0, 0, 0, 0);
        lB[lane][j + 0] = (_Float16)v4.x;
        lB[lane][j + 1] = (_Float16)v4.y;
        lB[lane][j + 2] = (_Float16)v4.z;
        lB[lane][j + 3] = (_Float16)v4.w;
      }
    }
    __syncthreads();

    // ---- fragments per CDNA5 WMMA VGPR layouts + MV*NCO WMMAs ----
    const int m16 = lane & 15;
    const int kb  = (lane < 16) ? 0 : 8;   // 16-bit A 16x32 layout
    const int kb2 = (lane < 16) ? 0 : 8;   // B rows striped across lanes
#pragma unroll
    for (int mv = 0; mv < MV; ++mv) {
      v16h a;
#pragma unroll
      for (int j = 0; j < 16; ++j) {
        const int r = j >> 1, h = j & 1;
        const int k = (r < 4) ? (kb + 2 * r + h) : (kb + 16 + 2 * (r - 4) + h);
        a[j] = lA[mv * 16 + m16][k];
      }
#pragma unroll
      for (int t = 0; t < NCO; ++t) {
        v16h b;
#pragma unroll
        for (int j = 0; j < 16; ++j) b[j] = lB[kb2 + j + ((j >> 3) << 3)][t * 16 + m16];
        acc[mv][t] = __builtin_amdgcn_wmma_f32_16x16x32_f16(
            false, a, false, b, (short)0, acc[mv][t], false, false);
      }
    }
    __syncthreads();
  }

  // ---- store D: lane n = cout lane, VGPR r -> M = r + 8*(lane>=16) ----
  const int co_n  = lane & 15;
  const int mbase = (lane >> 4) * 8;
#pragma unroll
  for (int mv = 0; mv < MV; ++mv)
#pragma unroll
    for (int t = 0; t < NCO; ++t) {
      const int co = co0 + t * 16 + co_n;
#pragma unroll
      for (int r = 0; r < 8; ++r) {
        const int vo = vbase + mv * 16 + mbase + r;
        if (vo < nvox) {
          float val = acc[mv][t][r];
          if (mask) val *= mask[vo];
          y[(size_t)vo * cout + co] = val;
        }
      }
    }
}

// ---------------------------------------------------------------------------
// Support kernels
// ---------------------------------------------------------------------------
__global__ void scatter_kernel(const float* __restrict__ feats,
                               const int* __restrict__ coords,
                               float* __restrict__ x0, float* __restrict__ mask1,
                               int npts, int cin, int H, int W) {
  const int p = blockIdx.x * blockDim.x + threadIdx.x;
  if (p >= npts) return;
  const int z = coords[p * 4 + 1];
  const int yy = coords[p * 4 + 2];
  const int xx = coords[p * 4 + 3];
  const int v = (z * H + yy) * W + xx;
  mask1[v] = 1.0f;
  for (int c = 0; c < cin; ++c) x0[(size_t)v * cin + c] = feats[p * cin + c];
}

__global__ void downmask_kernel(const float* __restrict__ min_,
                                float* __restrict__ mout,
                                int Di, int Hi, int Wi, int Do_, int Ho, int Wo,
                                int kd, int kh, int kw, int sd, int sh, int sw,
                                int pd, int ph, int pw) {
  const int v = blockIdx.x * blockDim.x + threadIdx.x;
  const int nvox = Do_ * Ho * Wo;
  if (v >= nvox) return;
  const int ow = v % Wo;
  const int oh = (v / Wo) % Ho;
  const int od = v / (Wo * Ho);
  float any = 0.0f;
  for (int td = 0; td < kd; ++td)
    for (int th = 0; th < kh; ++th)
      for (int tw = 0; tw < kw; ++tw) {
        const int id = od * sd + td - pd;
        const int ih = oh * sh + th - ph;
        const int iw = ow * sw + tw - pw;
        if ((unsigned)id < (unsigned)Di && (unsigned)ih < (unsigned)Hi &&
            (unsigned)iw < (unsigned)Wi)
          if (min_[(id * Hi + ih) * Wi + iw] > 0.0f) any = 1.0f;
      }
  mout[v] = any;
}

__global__ void mask_sum_kernel(const float* __restrict__ mask, int n,
                                float* __restrict__ dst) {
  __shared__ float sm[256];
  float s = 0.0f;
  for (int i = threadIdx.x; i < n; i += 256) s += mask[i];
  sm[threadIdx.x] = s;
  __syncthreads();
  for (int off = 128; off > 0; off >>= 1) {
    if (threadIdx.x < off) sm[threadIdx.x] += sm[threadIdx.x + off];
    __syncthreads();
  }
  if (threadIdx.x == 0) dst[0] = sm[0];
}

// one block per channel -> deterministic
__global__ void bn_stats_kernel(const float* __restrict__ y, int nvox, int C,
                                float* __restrict__ stats) {
  const int c = blockIdx.x;
  __shared__ float ssum[256], ssq[256];
  float s = 0.0f, q = 0.0f;
  for (int v = threadIdx.x; v < nvox; v += 256) {
    const float val = y[(size_t)v * C + c];
    s += val;
    q += val * val;
  }
  ssum[threadIdx.x] = s;
  ssq[threadIdx.x] = q;
  __syncthreads();
  for (int off = 128; off > 0; off >>= 1) {
    if (threadIdx.x < off) {
      ssum[threadIdx.x] += ssum[threadIdx.x + off];
      ssq[threadIdx.x] += ssq[threadIdx.x + off];
    }
    __syncthreads();
  }
  if (threadIdx.x == 0) {
    stats[c] = ssum[0];
    stats[C + c] = ssq[0];
  }
}

__global__ void bn_apply_kernel(const float* __restrict__ y,
                                const float* __restrict__ mask,
                                const float* __restrict__ stats,
                                const float* __restrict__ g,
                                const float* __restrict__ b,
                                const float* __restrict__ res,
                                float* __restrict__ out, int nvox, int C,
                                int relu, int resAdd) {
  const int i = blockIdx.x * blockDim.x + threadIdx.x;
  if (i >= nvox * C) return;
  const int v = i / C;
  const int c = i - v * C;
  const float n = fmaxf(stats[2 * C], 1.0f);
  const float mean = stats[c] / n;
  const float var = stats[C + c] / n - mean * mean;
  float val = (y[i] - mean) * g[c] * rsqrtf(var + BN_EPS) + b[c];
  if (resAdd) {
    val = fmaxf(val + res[i], 0.0f);
  } else if (relu) {
    val = fmaxf(val, 0.0f);
  }
  out[i] = val * mask[v];
}

__global__ void concat_kernel(const float* __restrict__ a,
                              const float* __restrict__ bb,
                              float* __restrict__ out, int nvox, int C) {
  const int i = blockIdx.x * blockDim.x + threadIdx.x;
  const int C2 = 2 * C;
  if (i >= nvox * C2) return;
  const int v = i / C2;
  const int c = i - v * C2;
  out[i] = (c < C) ? a[(size_t)v * C + c] : bb[(size_t)v * C + (c - C)];
}

// out = xm + pairwise channel reduce of xcat (reshape(cout,2).sum)
__global__ void chanred_add_kernel(const float* __restrict__ xm,
                                   const float* __restrict__ xcat,
                                   float* __restrict__ out, int nvox, int C) {
  const int i = blockIdx.x * blockDim.x + threadIdx.x;
  if (i >= nvox * C) return;
  const int v = i / C;
  const int c = i - v * C;
  out[i] = xm[i] + xcat[(size_t)v * 2 * C + 2 * c] +
           xcat[(size_t)v * 2 * C + 2 * c + 1];
}

__global__ void gather_kernel(const float* __restrict__ u1,
                              const int* __restrict__ coords,
                              float* __restrict__ out, int npts, int C, int H,
                              int W) {
  const int i = blockIdx.x * blockDim.x + threadIdx.x;
  if (i >= npts * C) return;
  const int p = i / C;
  const int c = i - p * C;
  const int z = coords[p * 4 + 1];
  const int yy = coords[p * 4 + 2];
  const int xx = coords[p * 4 + 3];
  out[i] = u1[((size_t)(z * H + yy) * W + xx) * C + c];
}

__global__ void to_nchw_kernel(const float* __restrict__ in,
                               float* __restrict__ out, int nvox, int C) {
  const int i = blockIdx.x * blockDim.x + threadIdx.x;
  if (i >= nvox * C) return;
  const int c = i / nvox;
  const int v = i - c * nvox;
  out[i] = in[(size_t)v * C + c];
}

// ---------------------------------------------------------------------------
// Host orchestration
// ---------------------------------------------------------------------------
namespace {

constexpr int D1 = 41, H1 = 100, W1 = 88;
constexpr int D2 = 21, H2 = 50, W2 = 44;
constexpr int D3 = 11, H3 = 25, W3 = 22;
constexpr int D4 = 5, H4 = 13, W4 = 11;
constexpr int D5 = 2, H5 = 13, W5 = 11;
constexpr int V1 = D1 * H1 * W1;
constexpr int V2 = D2 * H2 * W2;
constexpr int V3 = D3 * H3 * W3;
constexpr int V4 = D4 * H4 * W4;
constexpr int V5 = D5 * H5 * W5;
constexpr int NPTS = 20000;

inline int cdiv(int a, int b) { return (a + b - 1) / b; }

template <int CIN, int KD, int KH, int KW, int SD, int SH, int SW, int TR>
void conv_t(const float* x, const float* w, const float* mask, float* y,
            int Di, int Hi, int Wi, int Do_, int Ho, int Wo, int cout, int pd,
            int ph, int pw, hipStream_t s) {
  const int nvox = Do_ * Ho * Wo;
  if (cout >= 32) {
    dim3 grid(cdiv(nvox, 32), cout / 32);
    conv_wmma_kernel<CIN, KD, KH, KW, SD, SH, SW, TR, 2, 2><<<grid, 32, 0, s>>>(
        x, w, mask, y, Di, Hi, Wi, Do_, Ho, Wo, cout, pd, ph, pw);
  } else {
    dim3 grid(cdiv(nvox, 32), 1);
    conv_wmma_kernel<CIN, KD, KH, KW, SD, SH, SW, TR, 2, 1><<<grid, 32, 0, s>>>(
        x, w, mask, y, Di, Hi, Wi, Do_, Ho, Wo, cout, pd, ph, pw);
  }
}

// Runtime -> template dispatch over the (cin, kernel, stride, dir) shapes
// this network actually uses.
void conv_launch(const float* x, const float* w, const float* mask, float* y,
                 int Di, int Hi, int Wi, int Do_, int Ho, int Wo, int cin,
                 int cout, int kd, int kh, int kw, int sd, int sh, int sw,
                 int pd, int ph, int pw, int tr, hipStream_t s) {
  (void)kd; (void)kh; (void)kw;
  if (tr) {
    if (cin == 32)
      conv_t<32, 3, 3, 3, 2, 2, 2, 1>(x, w, mask, y, Di, Hi, Wi, Do_, Ho, Wo,
                                      cout, pd, ph, pw, s);
    else
      conv_t<64, 3, 3, 3, 2, 2, 2, 1>(x, w, mask, y, Di, Hi, Wi, Do_, Ho, Wo,
                                      cout, pd, ph, pw, s);
  } else if (kh == 1) {  // head conv k(3,1,1) s(2,1,1)
    conv_t<64, 3, 1, 1, 2, 1, 1, 0>(x, w, mask, y, Di, Hi, Wi, Do_, Ho, Wo,
                                    cout, pd, ph, pw, s);
  } else if (sd == 2) {  // downsampling spconv k3 s2
    if (cin == 16)
      conv_t<16, 3, 3, 3, 2, 2, 2, 0>(x, w, mask, y, Di, Hi, Wi, Do_, Ho, Wo,
                                      cout, pd, ph, pw, s);
    else if (cin == 32)
      conv_t<32, 3, 3, 3, 2, 2, 2, 0>(x, w, mask, y, Di, Hi, Wi, Do_, Ho, Wo,
                                      cout, pd, ph, pw, s);
    else
      conv_t<64, 3, 3, 3, 2, 2, 2, 0>(x, w, mask, y, Di, Hi, Wi, Do_, Ho, Wo,
                                      cout, pd, ph, pw, s);
  } else {  // submanifold k3 s1
    if (cin == 4)
      conv_t<4, 3, 3, 3, 1, 1, 1, 0>(x, w, mask, y, Di, Hi, Wi, Do_, Ho, Wo,
                                     cout, pd, ph, pw, s);
    else if (cin == 16)
      conv_t<16, 3, 3, 3, 1, 1, 1, 0>(x, w, mask, y, Di, Hi, Wi, Do_, Ho, Wo,
                                      cout, pd, ph, pw, s);
    else if (cin == 32)
      conv_t<32, 3, 3, 3, 1, 1, 1, 0>(x, w, mask, y, Di, Hi, Wi, Do_, Ho, Wo,
                                      cout, pd, ph, pw, s);
    else if (cin == 64)
      conv_t<64, 3, 3, 3, 1, 1, 1, 0>(x, w, mask, y, Di, Hi, Wi, Do_, Ho, Wo,
                                      cout, pd, ph, pw, s);
    else
      conv_t<128, 3, 3, 3, 1, 1, 1, 0>(x, w, mask, y, Di, Hi, Wi, Do_, Ho, Wo,
                                       cout, pd, ph, pw, s);
  }
}

void bn_launch(const float* y, const float* mask, const float* g,
               const float* b, const float* res, float* out, int nvox, int C,
               int relu, int resAdd, float* stats, hipStream_t s) {
  mask_sum_kernel<<<1, 256, 0, s>>>(mask, nvox, stats + 2 * C);
  bn_stats_kernel<<<C, 256, 0, s>>>(y, nvox, C, stats);
  bn_apply_kernel<<<cdiv(nvox * C, 256), 256, 0, s>>>(y, mask, stats, g, b,
                                                      res, out, nvox, C, relu,
                                                      resAdd);
}

struct Ctx {
  void* const* din;
  hipStream_t s;
  float* stats;
  float* ytmp;
  const float* Wt(int i) const { return (const float*)din[2 + 3 * i]; }
  const float* Gm(int i) const { return (const float*)din[3 + 3 * i]; }
  const float* Bt(int i) const { return (const float*)din[4 + 3 * i]; }

  // submanifold conv: same-pad 3x3x3, stride 1, + masked BN(/ReLU/residual)
  void subm(const float* x, int li, const float* mask, float* out, int Dd,
            int Hh, int Ww, int cin, int cout, int relu, const float* res,
            int resAdd) const {
    conv_launch(x, Wt(li), mask, ytmp, Dd, Hh, Ww, Dd, Hh, Ww, cin, cout, 3, 3,
                3, 1, 1, 1, 1, 1, 1, 0, s);
    bn_launch(ytmp, mask, Gm(li), Bt(li), res, out, Dd * Hh * Ww, cout, relu,
              resAdd, stats, s);
  }

  // ur_block: x_tr = basic_block(x_lat); xcat = [x_bot, x_tr];
  //           xm = subm(xcat); dest = xm + chan_reduce(xcat)
  void ur_block(const float* x_lat, const float* x_bot, const float* mask,
                int lt1, int lt2, int lm, float* s1, float* s2, float* scat,
                float* dest, int Dd, int Hh, int Ww, int C) const {
    const int nvox = Dd * Hh * Ww;
    subm(x_lat, lt1, mask, s1, Dd, Hh, Ww, C, C, 1, nullptr, 0);
    subm(s1, lt2, mask, s2, Dd, Hh, Ww, C, C, 0, x_lat, 1);  // +res, relu
    concat_kernel<<<cdiv(nvox * 2 * C, 256), 256, 0, s>>>(x_bot, s2, scat,
                                                          nvox, C);
    subm(scat, lm, mask, s1, Dd, Hh, Ww, 2 * C, C, 1, nullptr, 0);
    chanred_add_kernel<<<cdiv(nvox * C, 256), 256, 0, s>>>(s1, scat, dest,
                                                           nvox, C);
  }
};

}  // namespace

extern "C" void kernel_launch(void* const* d_in, const int* in_sizes, int n_in,
                              void* d_out, int out_size, void* d_ws,
                              size_t ws_size, hipStream_t stream) {
  (void)in_sizes; (void)n_in; (void)out_size; (void)ws_size;
  const float* feats = (const float*)d_in[0];
  const int* coords = (const int*)d_in[1];

  // ---- workspace arena ----
  char* ws = (char*)d_ws;
  size_t off = 0;
  auto alloc = [&](size_t nf) -> float* {
    float* p = (float*)(ws + off);
    off += ((nf * sizeof(float) + 255) / 256) * 256;
    return p;
  };
  float* stats = alloc(2 * 128 + 1);
  float* x0   = alloc((size_t)V1 * 4);
  float* mask1 = alloc(V1);
  float* ytmp = alloc((size_t)V1 * 16);
  float* xA   = alloc((size_t)V1 * 16);
  float* x1   = alloc((size_t)V1 * 16);
  float* mask2 = alloc(V2);
  float* mask3 = alloc(V3);
  float* mask4 = alloc(V4);
  float* mask5 = alloc(V5);
  float* x2a  = alloc((size_t)V2 * 32);
  float* x2b  = alloc((size_t)V2 * 32);
  float* x3a  = alloc((size_t)V3 * 64);
  float* x3b  = alloc((size_t)V3 * 64);
  float* x4a  = alloc((size_t)V4 * 64);
  float* x4b  = alloc((size_t)V4 * 64);
  float* outbuf = alloc((size_t)V5 * 128);
  float* s1   = alloc((size_t)V1 * 16);
  float* s2   = alloc((size_t)V1 * 16);
  float* scat = alloc((size_t)V1 * 32);
  float* uA   = alloc((size_t)V1 * 16);
  float* uB   = alloc((size_t)V1 * 16);

  Ctx c{d_in, stream, stats, ytmp};

  // ---- scatter input voxels to dense grid ----
  hipMemsetAsync(x0, 0, (size_t)V1 * 4 * sizeof(float), stream);
  hipMemsetAsync(mask1, 0, (size_t)V1 * sizeof(float), stream);
  scatter_kernel<<<cdiv(NPTS, 256), 256, 0, stream>>>(feats, coords, x0, mask1,
                                                      NPTS, 4, H1, W1);

  // ---- encoder ----
  c.subm(x0, 0, mask1, xA, D1, H1, W1, 4, 16, 1, nullptr, 0);   // inp
  c.subm(xA, 1, mask1, x1, D1, H1, W1, 16, 16, 1, nullptr, 0);  // c1

  downmask_kernel<<<cdiv(V2, 256), 256, 0, stream>>>(
      mask1, mask2, D1, H1, W1, D2, H2, W2, 3, 3, 3, 2, 2, 2, 1, 1, 1);
  conv_launch(x1, c.Wt(2), mask2, ytmp, D1, H1, W1, D2, H2, W2, 16, 32, 3, 3,
              3, 2, 2, 2, 1, 1, 1, 0, stream);                  // c2_0
  bn_launch(ytmp, mask2, c.Gm(2), c.Bt(2), nullptr, x2a, V2, 32, 1, 0, stats,
            stream);
  c.subm(x2a, 3, mask2, x2b, D2, H2, W2, 32, 32, 1, nullptr, 0);  // c2_1
  c.subm(x2b, 4, mask2, x2a, D2, H2, W2, 32, 32, 1, nullptr, 0);  // c2_2

  downmask_kernel<<<cdiv(V3, 256), 256, 0, stream>>>(
      mask2, mask3, D2, H2, W2, D3, H3, W3, 3, 3, 3, 2, 2, 2, 1, 1, 1);
  conv_launch(x2a, c.Wt(5), mask3, ytmp, D2, H2, W2, D3, H3, W3, 32, 64, 3, 3,
              3, 2, 2, 2, 1, 1, 1, 0, stream);                  // c3_0
  bn_launch(ytmp, mask3, c.Gm(5), c.Bt(5), nullptr, x3a, V3, 64, 1, 0, stats,
            stream);
  c.subm(x3a, 6, mask3, x3b, D3, H3, W3, 64, 64, 1, nullptr, 0);  // c3_1
  c.subm(x3b, 7, mask3, x3a, D3, H3, W3, 64, 64, 1, nullptr, 0);  // c3_2

  downmask_kernel<<<cdiv(V4, 256), 256, 0, stream>>>(
      mask3, mask4, D3, H3, W3, D4, H4, W4, 3, 3, 3, 2, 2, 2, 0, 1, 1);
  conv_launch(x3a, c.Wt(8), mask4, ytmp, D3, H3, W3, D4, H4, W4, 64, 64, 3, 3,
              3, 2, 2, 2, 0, 1, 1, 0, stream);                  // c4_0
  bn_launch(ytmp, mask4, c.Gm(8), c.Bt(8), nullptr, x4a, V4, 64, 1, 0, stats,
            stream);
  c.subm(x4a, 9, mask4, x4b, D4, H4, W4, 64, 64, 1, nullptr, 0);   // c4_1
  c.subm(x4b, 10, mask4, x4a, D4, H4, W4, 64, 64, 1, nullptr, 0);  // c4_2

  // ---- head: out = spconv(x4, k(3,1,1), s(2,1,1), p0) ----
  downmask_kernel<<<cdiv(V5, 256), 256, 0, stream>>>(
      mask4, mask5, D4, H4, W4, D5, H5, W5, 3, 1, 1, 2, 1, 1, 0, 0, 0);
  conv_launch(x4a, c.Wt(11), mask5, ytmp, D4, H4, W4, D5, H5, W5, 64, 128, 3,
              1, 1, 2, 1, 1, 0, 0, 0, 0, stream);               // out
  bn_launch(ytmp, mask5, c.Gm(11), c.Bt(11), nullptr, outbuf, V5, 128, 1, 0,
            stats, stream);

  // ---- decoder ----
  // L4 ur block (x_lat = x_bot = x4)
  c.ur_block(x4a, x4a, mask4, 12, 13, 14, s1, s2, scat, uA, D4, H4, W4, 64);
  // inv4: transposed conv L4 -> L3, k3 s2 pad(0,1,1)
  conv_launch(uA, c.Wt(15), mask3, ytmp, D4, H4, W4, D3, H3, W3, 64, 64, 3, 3,
              3, 2, 2, 2, 0, 1, 1, 1, stream);
  bn_launch(ytmp, mask3, c.Gm(15), c.Bt(15), nullptr, uB, V3, 64, 1, 0, stats,
            stream);

  c.ur_block(x3a, uB, mask3, 16, 17, 18, s1, s2, scat, uA, D3, H3, W3, 64);
  // inv3: L3 -> L2, 64->32, k3 s2 pad1
  conv_launch(uA, c.Wt(19), mask2, ytmp, D3, H3, W3, D2, H2, W2, 64, 32, 3, 3,
              3, 2, 2, 2, 1, 1, 1, 1, stream);
  bn_launch(ytmp, mask2, c.Gm(19), c.Bt(19), nullptr, uB, V2, 32, 1, 0, stats,
            stream);

  c.ur_block(x2a, uB, mask2, 20, 21, 22, s1, s2, scat, uA, D2, H2, W2, 32);
  // inv2: L2 -> L1, 32->16, k3 s2 pad1
  conv_launch(uA, c.Wt(23), mask1, ytmp, D2, H2, W2, D1, H1, W1, 32, 16, 3, 3,
              3, 2, 2, 2, 1, 1, 1, 1, stream);
  bn_launch(ytmp, mask1, c.Gm(23), c.Bt(23), nullptr, uB, V1, 16, 1, 0, stats,
            stream);

  c.ur_block(x1, uB, mask1, 24, 25, 26, s1, s2, scat, uA, D1, H1, W1, 16);
  c.subm(uA, 27, mask1, uB, D1, H1, W1, 16, 16, 1, nullptr, 0);  // c0

  // ---- outputs ----
  float* out_nchw = (float*)d_out;               // 128 * V5 floats (NCDHW)
  float* out_pts = out_nchw + (size_t)128 * V5;  // NPTS * 16 floats
  to_nchw_kernel<<<cdiv(V5 * 128, 256), 256, 0, stream>>>(outbuf, out_nchw, V5,
                                                          128);
  gather_kernel<<<cdiv(NPTS * 16, 256), 256, 0, stream>>>(uB, coords, out_pts,
                                                          NPTS, 16, H1, W1);
}
